// Router_15848429322274
// MI455X (gfx1250) — matrix operand order repaired and checked by reading, not compiled, not run on hardware
//
#include <hip/hip_runtime.h>
#include <hip/hip_bf16.h>

// Router MLP, transposed mapping, tiles processed sequentially:
//   h1^T = W1 x_norm^T,  h2^T = W2 h1^T,  score = w3 . h2
//  * W2's columns permuted at setup (psi) so layer-1's packed D dwords ARE
//    the layer-2 B operand (B2 = bit_cast(Q), zero data movement).
//  * One shared A1 set; tile-1's B1 operand is the LN dwords swapped down
//    via 3x v_permlanex16. No masking: A1's zero K-slots kill cross terms.
//  * ReLU after f16 pack -> v_pk_max_num_f16; layer-3 ReLU via
//    v_med3_f32(x, 0, FLT_MAX).
//  * 8 pairs per wave so the one-time weight-operand setup is amortized.

#define NUM_LAYERS 24
#define LN_EPS 1e-5f
#define WAVES_PER_BLOCK 8
#define FLT_BIG 3.402823466e+38f

typedef __attribute__((ext_vector_type(16))) _Float16 v16h;
typedef __attribute__((ext_vector_type(2)))  _Float16 h2v;
typedef __attribute__((ext_vector_type(8)))  float    v8f;
typedef __attribute__((ext_vector_type(8)))  int      v8i;
typedef __attribute__((ext_vector_type(2), aligned(4))) float f2a4;

__device__ __forceinline__ v8f wmma16x16x32(v16h a, v16h b, v8f c) {
  return __builtin_amdgcn_wmma_f32_16x16x32_f16(false, a, false, b, (short)0, c,
                                                false, false);
}
// Single-instruction ReLU that InstCombine keeps as v_med3_f32.
__device__ __forceinline__ float reluf(float v) {
  return __builtin_amdgcn_fmed3f(v, 0.0f, FLT_BIG);
}
// xor-16 half-wave swap in the VALU pipe.
__device__ __forceinline__ int pswap16(int v) {
  return __builtin_amdgcn_permlanex16(v, v, 0x76543210, 0xFEDCBA98, false, false);
}
__device__ __forceinline__ float pswap16f(float v) {
  return __builtin_bit_cast(float, pswap16(__builtin_bit_cast(int, v)));
}
__device__ __forceinline__ int pack_h2(float a, float b) {
  h2v p; p[0] = (_Float16)a; p[1] = (_Float16)b;
  return __builtin_bit_cast(int, p);
}
// W2 column permutation matching layer-1's natural D dword order.
__device__ __forceinline__ int psi(int k) {
  const int b = (k >> 3) & 3;
  return (b == 1 || b == 2) ? (k ^ 0x18) : k;
}

__global__ __launch_bounds__(256) void router_mlp_kernel(
    const float* __restrict__ x,
    const float* __restrict__ ln_w, const float* __restrict__ ln_b,
    const float* __restrict__ W1, const float* __restrict__ b1,
    const float* __restrict__ W2, const float* __restrict__ b2,
    const float* __restrict__ W3, const float* __restrict__ b3,
    float* __restrict__ out, int N)
{
  __shared__ float s_ln[NUM_LAYERS][8];   // [w0..w3, b0..b3]

  const int tid = threadIdx.x;
  if (tid < 96)       { s_ln[tid >> 2][tid & 3] = ln_w[tid]; }
  else if (tid < 192) { int k = tid - 96; s_ln[k >> 2][4 + (k & 3)] = ln_b[k]; }
  __syncthreads();

  const int  lane = tid & 31;
  const int  wid  = tid >> 5;
  const int  hl   = lane & 15;
  const bool lo   = lane < 16;

  // ---- Layer-1 weights (A-layout; W1 + bias at K=0..5, rest zero) --------
  v16h A1[4];
  #pragma unroll
  for (int c = 0; c < 4; ++c) {
    v16h a = {};
    if (lo) {                               // only K<8 nonzero -> low lanes
      const int j = 16 * c + hl;            // h1 feature (row of W1)
      #pragma unroll
      for (int k = 0; k < 5; ++k) a[k] = (_Float16)W1[j * 5 + k];
      a[5] = (_Float16)b1[j];               // bias lane of the GEMM
    }
    A1[c] = a;
  }
  // ---- Layer-2 weights: columns permuted by psi --------------------------
  v16h A2[2][2];
  #pragma unroll
  for (int m = 0; m < 2; ++m)
    #pragma unroll
    for (int kk = 0; kk < 2; ++kk) {
      const int j  = 16 * m + hl;           // h2 feature (row of W2)
      const int kb = 32 * kk + (lo ? 0 : 16);
      v16h a;
      #pragma unroll
      for (int h = 0; h < 8; ++h) a[h]     = (_Float16)W2[j * 64 + psi(kb + h)];
      #pragma unroll
      for (int h = 0; h < 8; ++h) a[8 + h] = (_Float16)W2[j * 64 + psi(kb + 8 + h)];
      A2[m][kk] = a;
    }
  // C/D layout: VGPR r -> M = (lo ? r : r+8); lane%16 = N (token).
  const int rb = lo ? 0 : 8;
  v8f C2[2];                                // layer-2 bias as WMMA C operand
  float w3v[2][8];
  #pragma unroll
  for (int m = 0; m < 2; ++m)
    #pragma unroll
    for (int r = 0; r < 8; ++r) {
      C2[m][r]  = b2[16 * m + rb + r];
      w3v[m][r] = W3[16 * m + rb + r];
    }
  const float bias3 = b3[0];
  const v8f zero8 = {};
  const h2v hzero = {};

  // B1 carriers live across iterations: dwords 3..7 are zeroed ONCE here and
  // never rewritten; only dwords 0..2 change per pair.
  v8i bx0 = {}, bx1 = {};

  const int NT     = (N + 15) >> 4;         // 16-token tiles
  const int NP     = (NT + 1) >> 1;         // tile pairs
  const int stride = gridDim.x * WAVES_PER_BLOCK;

  for (int pair = blockIdx.x * WAVES_PER_BLOCK + wid; pair < NP; pair += stride) {
    const int tb0 = pair * 32;
    const int t   = tb0 + lane;             // every lane owns one token
    const int tt  = t < N ? t : N - 1;

    // ---- LayerNorm for ALL 32 lanes (lane<16: tile0, lane>=16: tile1) ----
    const float* xp = x + (size_t)tt * 5;
    const f2a4 x01 = *(const f2a4*)xp;
    const f2a4 x23 = *(const f2a4*)(xp + 2);
    const float pos = xp[4];
    {
      const long nt = (long)t + (long)stride * 32;
      if (nt < N) __builtin_prefetch(x + (size_t)nt * 5, 0, 0);
    }
    const float mean = 0.25f * ((x01[0] + x01[1]) + (x23[0] + x23[1]));
    const float d0 = x01[0] - mean, d1 = x01[1] - mean;
    const float d2 = x23[0] - mean, d3 = x23[1] - mean;
    const float var  = 0.25f * (d0 * d0 + d1 * d1 + d2 * d2 + d3 * d3);
    const float rstd = rsqrtf(var + LN_EPS);
    int lid = (int)(pos * (float)NUM_LAYERS);
    lid = lid < 0 ? 0 : (lid > NUM_LAYERS - 1 ? NUM_LAYERS - 1 : lid);
    const float* lp = s_ln[lid];
    const int g0 = pack_h2(d0 * rstd * lp[0] + lp[4], d1 * rstd * lp[1] + lp[5]);
    const int g1 = pack_h2(d2 * rstd * lp[2] + lp[6], d3 * rstd * lp[3] + lp[7]);
    const int g2 = pack_h2(pos, 1.0f);      // pos + bias lane

    // ---- B1 operands, unmasked (A1's zero K-slots kill cross terms) ------
    bx0[0] = g0;          bx0[1] = g1;          bx0[2] = g2;
    bx1[0] = pswap16(g0); bx1[1] = pswap16(g1); bx1[2] = pswap16(g2);
    v16h Bx[2];
    Bx[0] = __builtin_bit_cast(v16h, bx0);
    Bx[1] = __builtin_bit_cast(v16h, bx1);

    // ---- Per tile: layer1 -> pack -> layer2 -> layer3 (low live range) ---
    float total[2];
    #pragma unroll
    for (int u = 0; u < 2; ++u) {
      // Layer 1: 4 independent WMMAs hide each other's D-read hazards.
      // D layout: lane = token, VGPR r = feature 16c+(lo?r:r+8).
      v8i Q0, Q1;                           // packed h1 == layer-2 B operand
      #pragma unroll
      for (int c = 0; c < 4; ++c) {
        v8f acc = wmma16x16x32(A1[c], Bx[u], zero8);
        #pragma unroll
        for (int i = 0; i < 4; ++i) {
          h2v p;
          p[0] = (_Float16)acc[2 * i];
          p[1] = (_Float16)acc[2 * i + 1];
          p = __builtin_elementwise_max(p, hzero);   // v_pk_max_num_f16
          const int d = __builtin_bit_cast(int, p);
          if (c < 2) Q0[4 * c + i] = d;
          else       Q1[4 * (c - 2) + i] = d;
        }
      }
      const v16h B20 = __builtin_bit_cast(v16h, Q0);
      const v16h B21 = __builtin_bit_cast(v16h, Q1);

      // Layer 2 (bias via C) + layer 3 dot, consumed immediately.
      float s0 = 0.0f, s1 = 0.0f;
      #pragma unroll
      for (int m = 0; m < 2; ++m) {
        v8f a   = wmma16x16x32(A2[m][0], B20, C2[m]);
        v8f acc = wmma16x16x32(A2[m][1], B21, a);
        #pragma unroll
        for (int r = 0; r < 8; ++r) {
          const float h = reluf(acc[r]);
          if (m == 0) s0 = fmaf(h, w3v[0][r], s0);
          else        s1 = fmaf(h, w3v[1][r], s1);
        }
      }
      const float partial = s0 + s1;
      total[u] = partial + pswap16f(partial);
    }
    // lane<16 holds tile0 token lane; lane>=16 holds tile1 token lane-16.
    if (t < N)
      __builtin_nontemporal_store((lo ? total[0] : total[1]) + bias3, out + t);
  }
}

extern "C" void kernel_launch(void* const* d_in, const int* in_sizes, int n_in,
                              void* d_out, int out_size, void* d_ws, size_t ws_size,
                              hipStream_t stream) {
  const float* x    = (const float*)d_in[0];
  const float* ln_w = (const float*)d_in[1];
  const float* ln_b = (const float*)d_in[2];
  const float* W1   = (const float*)d_in[3];
  const float* b1   = (const float*)d_in[4];
  const float* W2   = (const float*)d_in[5];
  const float* b2   = (const float*)d_in[6];
  const float* W3   = (const float*)d_in[7];
  const float* b3   = (const float*)d_in[8];
  float* out = (float*)d_out;

  const int N  = in_sizes[0] / 5;          // 2,097,152 tokens
  const int NT = (N + 15) / 16;            // 16-token tiles
  const int NP = (NT + 1) / 2;             // tile pairs per wave-iteration
  // 1024 blocks x 8 waves = 8192 waves -> ~8 pairs/wave: one-time weight
  // setup (~250 instrs incl. ~90 loads) amortizes to ~10% instead of ~40%.
  int blocks = (NP + WAVES_PER_BLOCK - 1) / WAVES_PER_BLOCK;
  if (blocks > 1024) blocks = 1024;
  if (blocks < 1) blocks = 1;

  router_mlp_kernel<<<blocks, 32 * WAVES_PER_BLOCK, 0, stream>>>(
      x, ln_w, ln_b, W1, b1, W2, b2, W3, b3, out, N);
}
